// EnforcedLSTMModel_2388001817339
// MI455X (gfx1250) — compile-verified, matrix operand order checked by reading, not run on hardware
//
#include <hip/hip_runtime.h>

typedef __attribute__((ext_vector_type(16))) __bf16 v16bf;
typedef __attribute__((ext_vector_type(8)))  __bf16 v8bf;
typedef __attribute__((ext_vector_type(8)))  float  v8f;

#define H    512
#define BB   512      // batch
#define TOUT 64       // output size
#define FD   64       // forcing dim
#define IND  128      // lstm0 input dim
#define G4H  2048     // 4*H
#define NBLOCKS  128
#define NTHREADS 256
#define KC   128      // K-chunk staged through LDS
#define RS   136      // padded LDS row stride (elements) -> bank spread

typedef __bf16 SmemBuf[4][16][RS];   // [gate][row][k], 4*16*136*2 = 17408 B

// ---------------- helpers ----------------

__device__ __forceinline__ v16bf ld_frag(const __bf16* p) {
    // CDNA5 16-bit A/B fragment: this lane's contiguous K-octets live at p[0..7]
    // and p[16..23] (the +8/+24 octet select is folded into p via koff8).
    union { v16bf v; v8bf h[2]; } u;
    u.h[0] = *(const v8bf*)(p);
    u.h[1] = *(const v8bf*)(p + 16);
    return u.v;
}

__device__ __forceinline__ v8f wmma_bf16(v16bf a, v16bf b, v8f c) {
    return __builtin_amdgcn_wmma_f32_16x16x32_bf16(
        false, a, false, b, (short)0, c, false, false);
}

__device__ __forceinline__ float sig_(float x) { return 1.0f / (1.0f + __expf(-x)); }

__device__ __forceinline__ v8f bias_bcast(const float* bias, int idx) {
    float bv = bias[idx];
    union { v8f v; float f[8]; } u;
#pragma unroll
    for (int i = 0; i < 8; ++i) u.f[i] = bv;
    return u.v;
}

// ---- B staging: 256 threads cooperatively move 4x16x128 bf16 (16 KB) ----
// c = tid + it*256 -> gate q = c>>8, row = (c&255)>>4, k-octet = (c&15)*8
__device__ __forceinline__ void stage_load(const __bf16* __restrict__ W, int K,
                                           int k0, int jt16, int tid, v8bf g[4]) {
#pragma unroll
    for (int it = 0; it < 4; ++it) {
        int c = tid + it * 256;
        int q = c >> 8, rem = c & 255;
        int row = rem >> 4, kc8 = (rem & 15) * 8;
        g[it] = *(const v8bf*)(W + (size_t)(q * 512 + jt16 + row) * K + k0 + kc8);
    }
}

__device__ __forceinline__ void stage_store(SmemBuf& buf, int tid, const v8bf g[4]) {
#pragma unroll
    for (int it = 0; it < 4; ++it) {
        int c = tid + it * 256;
        int q = c >> 8, rem = c & 255;
        int row = rem >> 4, kc8 = (rem & 15) * 8;
        *(v8bf*)(&buf[q][row][kc8]) = g[it];
    }
}

// ---- one KC-chunk of gates GEMM: A from global, B from LDS ----
// B fragments are software-pipelined: the ds_load for WMMA i+1 issues before
// WMMA i, so each s_wait_dscnt hides behind a full WMMA of execution.
__device__ __forceinline__ void compute_chunk(v8f acc[4],
                                              const __bf16* __restrict__ X, int K,
                                              int arow_n, int xk0,
                                              const SmemBuf& buf,
                                              int nlane, int koff8) {
    const __bf16* xp = X + (size_t)arow_n * K + xk0 + koff8;
    v16bf a[4];
#pragma unroll
    for (int ks = 0; ks < 4; ++ks) a[ks] = ld_frag(xp + ks * 32);

    v16bf bcur = ld_frag(&buf[0][nlane][koff8]);
#pragma unroll
    for (int i = 0; i < 16; ++i) {
        const int ks = i >> 2, q = i & 3;
        v16bf bnext;
        if (i < 15) {
            const int ni = i + 1, nks = ni >> 2, nq = ni & 3;
            bnext = ld_frag(&buf[nq][nlane][nks * 32 + koff8]);
        }
        acc[q] = wmma_bf16(a[ks], bcur, acc[q]);
        if (i < 15) bcur = bnext;
    }
}

// ---- full gate GEMM for one layer: two (X,W,K) segments, LDS double-buffer ----
template <int NCH0, int NCH1>
__device__ __forceinline__ void layer_pass(v8f acc[4],
        const __bf16* __restrict__ X0, const __bf16* __restrict__ W0, int K0,
        const __bf16* __restrict__ X1, const __bf16* __restrict__ W1, int K1,
        SmemBuf* smem, int tid, int jt16, int arow_n, int nlane, int koff8) {
    v8bf g[4];
    stage_load(W0, K0, 0, jt16, tid, g);
    stage_store(smem[0], tid, g);
    __syncthreads();
    const int total = NCH0 + NCH1;
#pragma unroll 1
    for (int c = 0; c < total; ++c) {
        const int nc = c + 1;
        if (nc < total) {
            const __bf16* W = (nc < NCH0) ? W0 : W1;
            int K  = (nc < NCH0) ? K0 : K1;
            int k0 = ((nc < NCH0) ? nc : nc - NCH0) * KC;
            stage_load(W, K, k0, jt16, tid, g);   // global loads overlap WMMAs below
        }
        {
            const __bf16* X = (c < NCH0) ? X0 : X1;
            int K  = (c < NCH0) ? K0 : K1;
            int k0 = ((c < NCH0) ? c : c - NCH0) * KC;
            compute_chunk(acc, X, K, arow_n, k0, smem[c & 1], nlane, koff8);
        }
        if (nc < total) stage_store(smem[nc & 1], tid, g);
        __syncthreads();
    }
}

// ---- small GEMM with B straight from global (FC layer, fcw is tiny) ----
__device__ __forceinline__ void gemm_fc(v8f& acc,
                                        const __bf16* __restrict__ X,
                                        const __bf16* __restrict__ W,
                                        int K, int arow_n, int gcol,
                                        int nlane, int koff8) {
    const __bf16* xp = X + (size_t)arow_n * K + koff8;
    const __bf16* wp = W + (size_t)(gcol + nlane) * K + koff8;
#pragma unroll 1
    for (int k0 = 0; k0 < K; k0 += 32) {
        v16bf a = ld_frag(xp + k0);
        v16bf b = ld_frag(wp + k0);
        acc = wmma_bf16(a, b, acc);
    }
}

// i,f,g,o -> (c,h) update; h tile stored bf16 for the next GEMM's A operand.
__device__ __forceinline__ void lstm_update(v8f acc[4], v8f& c,
                                            __bf16* __restrict__ hbuf,
                                            int btile, int jt16,
                                            int m_half, int nlane) {
    union U { v8f v; float f[8]; };
    U ai, af, ag, ao, uc;
    ai.v = acc[0]; af.v = acc[1]; ag.v = acc[2]; ao.v = acc[3]; uc.v = c;
    const int col = jt16 + nlane;
#pragma unroll
    for (int v = 0; v < 8; ++v) {
        float iv = sig_(ai.f[v]);
        float fv = sig_(af.f[v]);
        float gv = tanhf(ag.f[v]);
        float ov = sig_(ao.f[v]);
        float cn = fv * uc.f[v] + iv * gv;
        uc.f[v] = cn;
        float hv = ov * tanhf(cn);
        int row = btile * 16 + m_half * 8 + v;
        hbuf[(size_t)row * H + col] = (__bf16)hv;
    }
    c = uc.v;
}

// device-wide sense-reversing barrier (persistent kernel; 128 co-resident WGs)
__device__ __forceinline__ void grid_sync(unsigned* bar) {
    __syncthreads();
    if (threadIdx.x == 0) {
        __threadfence();
        unsigned gen = __hip_atomic_load(&bar[1], __ATOMIC_RELAXED, __HIP_MEMORY_SCOPE_AGENT);
        unsigned arr = __hip_atomic_fetch_add(&bar[0], 1u, __ATOMIC_ACQ_REL, __HIP_MEMORY_SCOPE_AGENT);
        if (arr == (unsigned)(NBLOCKS - 1)) {
            __hip_atomic_store(&bar[0], 0u, __ATOMIC_RELAXED, __HIP_MEMORY_SCOPE_AGENT);
            __hip_atomic_fetch_add(&bar[1], 1u, __ATOMIC_RELEASE, __HIP_MEMORY_SCOPE_AGENT);
        } else {
            while (__hip_atomic_load(&bar[1], __ATOMIC_ACQUIRE, __HIP_MEMORY_SCOPE_AGENT) == gen)
                __builtin_amdgcn_s_sleep(2);
        }
        __threadfence();
    }
    __syncthreads();
}

// ---------------- init kernel ----------------

__global__ void lstm_init_kernel(
    const float* __restrict__ inputs, const float* __restrict__ forcing,
    const float* __restrict__ Wih0f, const float* __restrict__ Whh0f,
    const float* __restrict__ bih0,  const float* __restrict__ bhh0,
    const float* __restrict__ Wih1f, const float* __restrict__ Whh1f,
    const float* __restrict__ bih1,  const float* __restrict__ bhh1,
    const float* __restrict__ fcwf,  const float* __restrict__ fcbf,
    __bf16* Wih0, __bf16* Whh0, __bf16* Wih1, __bf16* Whh1, __bf16* fcw,
    float* bias0, float* bias1, float* fcb,
    __bf16* xin, __bf16* h0, __bf16* h1, float* out, unsigned* bar) {
    int tid = blockIdx.x * blockDim.x + threadIdx.x;
    int stride = gridDim.x * blockDim.x;
    for (int i = tid; i < G4H * IND; i += stride) Wih0[i] = (__bf16)Wih0f[i];
    for (int i = tid; i < G4H * H; i += stride) {
        Whh0[i] = (__bf16)Whh0f[i];
        Wih1[i] = (__bf16)Wih1f[i];
        Whh1[i] = (__bf16)Whh1f[i];
    }
    for (int i = tid; i < TOUT * H; i += stride) fcw[i] = (__bf16)fcwf[i];
    for (int i = tid; i < G4H; i += stride) {
        bias0[i] = bih0[i] + bhh0[i];
        bias1[i] = bih1[i] + bhh1[i];
    }
    for (int i = tid; i < TOUT; i += stride) fcb[i] = fcbf[i];
    for (int i = tid; i < BB * H; i += stride) { h0[i] = (__bf16)0.0f; h1[i] = (__bf16)0.0f; }
    for (int i = tid; i < BB * TOUT; i += stride) out[i] = inputs[i];           // out[0] = inputs
    for (int i = tid; i < BB * IND; i += stride) {                              // xin = [x0, forcing[0]]
        int r = i >> 7, c = i & 127;
        float v = (c < TOUT) ? inputs[r * TOUT + c] : forcing[r * FD + (c - TOUT)];
        xin[i] = (__bf16)v;
    }
    if (tid == 0) { bar[0] = 0u; bar[1] = 0u; }
}

// ---------------- persistent recurrence kernel ----------------
// block = (jtile, btile-group): all 8 waves share jtile -> B staged once in LDS.

__global__ __launch_bounds__(NTHREADS, 1) void lstm_persistent_kernel(
    const float* __restrict__ forcing, float* __restrict__ out,
    const __bf16* __restrict__ Wih0, const __bf16* __restrict__ Whh0,
    const __bf16* __restrict__ Wih1, const __bf16* __restrict__ Whh1,
    const __bf16* __restrict__ fcw,
    const float* __restrict__ bias0, const float* __restrict__ bias1,
    const float* __restrict__ fcb,
    __bf16* __restrict__ xin, __bf16* __restrict__ h0buf, __bf16* __restrict__ h1buf,
    unsigned* bar, int T) {
    __shared__ SmemBuf smem[2];

    const int tid    = threadIdx.x;
    const int lane   = tid & 31;
    const int wave   = tid >> 5;
    const int jtile  = blockIdx.x >> 2;              // 0..31 (hidden tile, per block)
    const int btile  = (blockIdx.x & 3) * 8 + wave;  // 0..31 (batch tile, per wave)
    const int m_half = lane >> 4;                    // 0|1
    const int koff8  = m_half << 3;                  // K-octet select
    const int nlane  = lane & 15;                    // tile column
    const int jt16   = jtile * 16;
    const int arow_n = btile * 16 + nlane;

    v8f c0 = {};  // cell state tiles live in VGPRs for the whole rollout
    v8f c1 = {};

#pragma unroll 1
    for (int t = 0; t < T - 1; ++t) {
        // ---- layer 0: gates = xin*Wih0^T + h0*Whh0^T + b ----
        {
            v8f acc[4];
#pragma unroll
            for (int q = 0; q < 4; ++q) acc[q] = bias_bcast(bias0, q * H + jt16 + nlane);
            layer_pass<IND / KC, H / KC>(acc, xin, Wih0, IND, h0buf, Whh0, H,
                                         smem, tid, jt16, arow_n, nlane, koff8);
            lstm_update(acc, c0, h0buf, btile, jt16, m_half, nlane);
        }
        grid_sync(bar);

        // ---- layer 1: gates = h0*Wih1^T + h1*Whh1^T + b ----
        {
            v8f acc[4];
#pragma unroll
            for (int q = 0; q < 4; ++q) acc[q] = bias_bcast(bias1, q * H + jt16 + nlane);
            layer_pass<H / KC, H / KC>(acc, h0buf, Wih1, H, h1buf, Whh1, H,
                                       smem, tid, jt16, arow_n, nlane, koff8);
            lstm_update(acc, c1, h1buf, btile, jt16, m_half, nlane);
        }
        grid_sync(bar);

        // ---- FC + output + next-step input assembly ----
        if (jtile < 4) {
            // x_new tile (btile, jtile): 512x64 = 32 btiles x 4 col-tiles
            v8f acc = bias_bcast(fcb, jt16 + nlane);
            gemm_fc(acc, h1buf, fcw, H, arow_n, jt16, nlane, koff8);
            union { v8f v; float f[8]; } u; u.v = acc;
            const int col = jt16 + nlane;  // 0..63
#pragma unroll
            for (int v = 0; v < 8; ++v) {
                int row = btile * 16 + m_half * 8 + v;
                float xv = u.f[v];
                out[((size_t)(t + 1) * BB + row) * TOUT + col] = xv;   // outputs[t+1]
                xin[(size_t)row * IND + col] = (__bf16)xv;             // next xin[:, :64]
            }
        } else if (jtile < 8) {
            // next xin[:, 64:128] = forcing[t+1]
            const int col = (jtile - 4) * 16 + nlane;  // 0..63
#pragma unroll
            for (int v = 0; v < 8; ++v) {
                int row = btile * 16 + m_half * 8 + v;
                float fv = forcing[((size_t)(t + 1) * BB + row) * FD + col];
                xin[(size_t)row * IND + TOUT + col] = (__bf16)fv;
            }
        }
        grid_sync(bar);
    }
}

// ---------------- launch ----------------

extern "C" void kernel_launch(void* const* d_in, const int* in_sizes, int n_in,
                              void* d_out, int out_size, void* d_ws, size_t ws_size,
                              hipStream_t stream) {
    (void)in_sizes; (void)n_in; (void)ws_size;
    const float* inputs  = (const float*)d_in[0];
    const float* forcing = (const float*)d_in[1];
    // d_in[2] = timespan (device scalar); derive T from out_size instead.
    const float* Wih0f = (const float*)d_in[3];
    const float* Whh0f = (const float*)d_in[4];
    const float* bih0  = (const float*)d_in[5];
    const float* bhh0  = (const float*)d_in[6];
    const float* Wih1f = (const float*)d_in[7];
    const float* Whh1f = (const float*)d_in[8];
    const float* bih1  = (const float*)d_in[9];
    const float* bhh1  = (const float*)d_in[10];
    const float* fcwf  = (const float*)d_in[11];
    const float* fcbf  = (const float*)d_in[12];
    float* out = (float*)d_out;
    const int T = out_size / (BB * TOUT);

    // workspace layout (256B-aligned slabs)
    char* ws = (char*)d_ws;
    size_t off = 0;
    auto take = [&](size_t bytes) { void* p = ws + off; off += (bytes + 255) & ~(size_t)255; return p; };
    __bf16* Wih0 = (__bf16*)take((size_t)G4H * IND * 2);
    __bf16* Whh0 = (__bf16*)take((size_t)G4H * H * 2);
    __bf16* Wih1 = (__bf16*)take((size_t)G4H * H * 2);
    __bf16* Whh1 = (__bf16*)take((size_t)G4H * H * 2);
    __bf16* fcw  = (__bf16*)take((size_t)TOUT * H * 2);
    float* bias0 = (float*)take((size_t)G4H * 4);
    float* bias1 = (float*)take((size_t)G4H * 4);
    float* fcb   = (float*)take((size_t)TOUT * 4);
    __bf16* xin  = (__bf16*)take((size_t)BB * IND * 2);
    __bf16* h0b  = (__bf16*)take((size_t)BB * H * 2);
    __bf16* h1b  = (__bf16*)take((size_t)BB * H * 2);
    unsigned* bar = (unsigned*)take(256);

    lstm_init_kernel<<<512, 256, 0, stream>>>(
        inputs, forcing, Wih0f, Whh0f, bih0, bhh0, Wih1f, Whh1f, bih1, bhh1,
        fcwf, fcbf, Wih0, Whh0, Wih1, Whh1, fcw, bias0, bias1, fcb,
        xin, h0b, h1b, out, bar);

    lstm_persistent_kernel<<<NBLOCKS, NTHREADS, 0, stream>>>(
        forcing, out, Wih0, Whh0, Wih1, Whh1, fcw, bias0, bias1, fcb,
        xin, h0b, h1b, bar, T);
}
